// NeXtVLAD_20444044329187
// MI455X (gfx1250) — matrix operand order repaired and checked by reading, not compiled, not run on hardware
//
#include <hip/hip_runtime.h>
#include <hip/hip_bf16.h>
#include <math.h>

typedef __attribute__((ext_vector_type(16))) _Float16 v16h;
typedef __attribute__((ext_vector_type(8)))  _Float16 v8h;
typedef __attribute__((ext_vector_type(4)))  _Float16 v4h;
typedef __attribute__((ext_vector_type(8)))  float    v8f;

#define M_TILE 32
#define D_IN   1024
#define D2     2048
#define GRP    8
#define NC     32
#define ND     256
#define MSEQ   512
#define NB     32
#define NTILES (MSEQ / M_TILE)   // 16

// ---- workspace layout (bytes) ----
#define WS_VPART 0                         // 16*32*256 f32 = 524288
#define WS_WE16  524288                    // 2048*1024 f16 = 4 MB
#define WS_W116  (524288 + 4194304)        // 256*2048  f16 = 1 MB
#define WS_TOTAL (524288 + 4194304 + 1048576)

// ---- LDS layout (bytes) ----
#define OFF_XN   0                 // 32*1024 f16  = 65536
#define OFF_XE   65536             // 32*2048 f16  = 131072
#define OFF_L1   196608            // 32*256  f32  = 32768   (phase1: PART aliases here)
#define OFF_A2   229376            // 32*8    f32  = 1024
#define OFF_WGK  230400            // 256     f32  = 1024
#define OFF_WK   231424            // 32      f32  = 128     (phase1: RNORM aliases here)
#define LDS_BYTES 231552

static_assert(LDS_BYTES <= 320 * 1024, "exceeds WGP LDS");

// D = A(16x32 f16) * B(32x16 f16) + C(16x16 f32)
__device__ __forceinline__ v8f wmma16(v16h a, v16h b, v8f c) {
  return __builtin_amdgcn_wmma_f32_16x16x32_f16(false, a, false, b, (short)0, c,
                                                false, false);
}

__device__ __forceinline__ v16h cat8(v8h lo, v8h hi) {
  return __builtin_shufflevector(lo, hi, 0, 1, 2, 3, 4, 5, 6, 7,
                                 8, 9, 10, 11, 12, 13, 14, 15);
}

// A fragment (16x32 f16, row-major f16 source, leading dim `ld`), ISA layout:
// lane L: row = L&15, Kbase = (L>=16 ? 8 : 0); h[0..7] = Kbase+0..7,
// h[8..15] = Kbase+16..23.  Two aligned 16B chunks -> ds_load_b128 x2.
__device__ __forceinline__ v16h load_a_lds(const _Float16* base, int ld, int lane,
                                           int k0) {
  const _Float16* p = base + (lane & 15) * ld + k0 + ((lane >> 4) << 3);
  return cat8(*(const v8h*)p, *(const v8h*)(p + 16));
}

// B fragment (32x16 f16) from row-major f16 weights W16[n][k] (leading dim `ld`):
// lane L holds column n0+(L&15); K = (L>=16?8:0)+i, i<8; +16 for i>=8.  Two 8-half chunks
// -> global_load_b128 x2, no converts.
__device__ __forceinline__ v16h load_b_h(const _Float16* __restrict__ W16, int ld,
                                         int lane, int n0, int k0) {
  const _Float16* p = W16 + (size_t)(n0 + (lane & 15)) * ld + k0 + ((lane >> 4) << 3);
  return cat8(*(const v8h*)p, *(const v8h*)(p + 16));
}

// One-time (per launch) f32 -> f16 weight conversion; bandwidth-bound, idempotent.
__global__ __launch_bounds__(256) void cvt_f32_to_f16(const float* __restrict__ src,
                                                      _Float16* __restrict__ dst,
                                                      int n4) {
  const int i = blockIdx.x * 256 + threadIdx.x;
  if (i < n4) {
    const float4 v = ((const float4*)src)[i];
    v4h h;
    h[0] = (_Float16)v.x; h[1] = (_Float16)v.y;
    h[2] = (_Float16)v.z; h[3] = (_Float16)v.w;
    *(v4h*)(dst + 4 * (size_t)i) = h;
  }
}

__global__ __launch_bounds__(256) void nextvlad_main(
    const float* __restrict__ x, const _Float16* __restrict__ We16,
    const float* __restrict__ be, const _Float16* __restrict__ W116,
    const float* __restrict__ b1, const float* __restrict__ W2,
    const float* __restrict__ b2, const float* __restrict__ cen,
    float* __restrict__ vpart) {
  extern __shared__ char smem[];
  _Float16* XN = (_Float16*)(smem + OFF_XN);
  _Float16* XE = (_Float16*)(smem + OFF_XE);
  float* L1  = (float*)(smem + OFF_L1);
  float* A2  = (float*)(smem + OFF_A2);
  float* WGK = (float*)(smem + OFF_WGK);
  float* WK  = (float*)(smem + OFF_WK);
  float* PART  = L1;   // phase-1 scratch (dead before L1 is written)
  float* RNORM = WK;   // phase-1 scratch (dead before WK is written)

  const int t = threadIdx.x;
  const int lane = t & 31;
  const int w = t >> 5;              // wave id 0..7
  const int tile = blockIdx.x;       // m-tile 0..15
  const int b = blockIdx.y;          // batch 0..31
  const float* xtile = x + ((size_t)b * MSEQ + tile * M_TILE) * D_IN;

  // ---------- Phase 1: row L2-norms, store normalized tile as f16 ----------
  {
    const int r = t >> 3, j = t & 7;
    const float* p = xtile + r * D_IN + j * 128;
    float s = 0.f;
#pragma unroll 4
    for (int i = 0; i < 128; ++i) { float v = p[i]; s += v * v; }
    PART[t] = s;
  }
  __syncthreads();
  if (t < M_TILE) {
    float s = 0.f;
    for (int j = 0; j < 8; ++j) s += PART[t * 8 + j];
    RNORM[t] = 1.0f / fmaxf(sqrtf(s), 1e-12f);
  }
  __syncthreads();
  for (int i = 0; i < 128; ++i) {
    int flat = t + 256 * i;              // coalesced
    XN[flat] = (_Float16)(xtile[flat] * RNORM[flat >> 10]);
  }
  __syncthreads();

  // ---------- Phase 2: XE = XN @ We^T + be  (WMMA f16 -> f32) ----------
  {
    const int msub = w >> 2;                 // 16-row subtile 0..1
    const int ech = w & 3;                   // 512-wide e chunk 0..3
    const _Float16* Abase = XN + msub * 16 * D_IN;
    const int rbase = msub * 16 + ((lane >> 4) << 3);
    const int cl = lane & 15;
    for (int eg = 0; eg < 8; ++eg) {         // 8 groups of 4 e-tiles
      const int e0 = ech * 512 + eg * 64;
      v8f acc0 = {}, acc1 = {}, acc2 = {}, acc3 = {};
      for (int k0 = 0; k0 < D_IN; k0 += 32) {
        v16h a = load_a_lds(Abase, D_IN, lane, k0);
        acc0 = wmma16(a, load_b_h(We16, D_IN, lane, e0,      k0), acc0);
        acc1 = wmma16(a, load_b_h(We16, D_IN, lane, e0 + 16, k0), acc1);
        acc2 = wmma16(a, load_b_h(We16, D_IN, lane, e0 + 32, k0), acc2);
        acc3 = wmma16(a, load_b_h(We16, D_IN, lane, e0 + 48, k0), acc3);
      }
      v8f accs[4] = {acc0, acc1, acc2, acc3};
#pragma unroll
      for (int tt = 0; tt < 4; ++tt) {
        const int col = e0 + tt * 16 + cl;
        const float bias = be[col];
#pragma unroll
        for (int r = 0; r < 8; ++r)
          XE[(rbase + r) * D2 + col] = (_Float16)(accs[tt][r] + bias);
      }
    }
  }
  __syncthreads();

  // ---------- Phase 3: logits1 = XE @ W1^T + b1 -> L1 (f32) ----------
  {
    const int msub = w >> 2;
    const int n0 = (w & 3) * 64;
    const _Float16* Abase = XE + msub * 16 * D2;
    const int rbase = msub * 16 + ((lane >> 4) << 3);
    const int cl = lane & 15;
    v8f acc0 = {}, acc1 = {}, acc2 = {}, acc3 = {};
    for (int k0 = 0; k0 < D2; k0 += 32) {
      v16h a = load_a_lds(Abase, D2, lane, k0);
      acc0 = wmma16(a, load_b_h(W116, D2, lane, n0,      k0), acc0);
      acc1 = wmma16(a, load_b_h(W116, D2, lane, n0 + 16, k0), acc1);
      acc2 = wmma16(a, load_b_h(W116, D2, lane, n0 + 32, k0), acc2);
      acc3 = wmma16(a, load_b_h(W116, D2, lane, n0 + 48, k0), acc3);
    }
    v8f accs[4] = {acc0, acc1, acc2, acc3};
#pragma unroll
    for (int tt = 0; tt < 4; ++tt) {
      const int col = n0 + tt * 16 + cl;
      const float bias = b1[col];
#pragma unroll
      for (int r = 0; r < 8; ++r)
        L1[(rbase + r) * ND + col] = accs[tt][r] + bias;
    }
  }
  // ---------- Phase 3b: logits2 (N=8, VALU, 8-wide) -> A2 (raw) ----------
  {
    const int r = t >> 3, g = t & 7;
    const v8h* xr8 = (const v8h*)(XE + r * D2);
    const float* w2 = W2 + g * D2;
    float s = b2[g];
    for (int e8 = 0; e8 < D2 / 8; ++e8) {
      const v8h xv = xr8[e8];
      const float* wp = w2 + e8 * 8;
#pragma unroll
      for (int i = 0; i < 8; ++i) s += (float)xv[i] * wp[i];
    }
    A2[r * GRP + g] = s;
  }
  __syncthreads();

  // ---------- Phase 4: softmaxes ----------
  {  // a1 over NC, in place on L1 (one thread per (row, group))
    const int r = t >> 3, g = t & 7;
    float* p = L1 + r * ND + g * NC;
    float mx = p[0];
    for (int k = 1; k < NC; ++k) mx = fmaxf(mx, p[k]);
    float s = 0.f;
    for (int k = 0; k < NC; ++k) { float e = __expf(p[k] - mx); p[k] = e; s += e; }
    const float inv = 1.0f / s;
    for (int k = 0; k < NC; ++k) p[k] *= inv;
  }
  if (t < M_TILE) {  // a2 over G
    float l[GRP];
    float mx = -1e30f;
    for (int g = 0; g < GRP; ++g) { l[g] = A2[t * GRP + g]; mx = fmaxf(mx, l[g]); }
    float s = 0.f;
    for (int g = 0; g < GRP; ++g) { l[g] = __expf(l[g] - mx); s += l[g]; }
    const float inv = 1.0f / s;
    for (int g = 0; g < GRP; ++g) A2[t * GRP + g] = l[g] * inv;
  }
  __syncthreads();

  // ---------- Phase 5: factored aggregation ----------
  {  // WGK[g,k] = sum_r a2[r,g] * a1[r,g,k]
    const int g = t >> 5, k = t & 31;
    float s = 0.f;
    for (int r = 0; r < M_TILE; ++r)
      s += A2[r * GRP + g] * L1[r * ND + g * NC + k];
    WGK[t] = s;
  }
  __syncthreads();
  if (t < NC) {  // WK[k] = sum_g WGK[g,k]
    float s = 0.f;
    for (int g = 0; g < GRP; ++g) s += WGK[g * NC + t];
    WK[t] = s;
  }
  __syncthreads();
  {  // vpart[tile,b,n] = sum_{r,g} a2*xg  -  sum_k WK[k]*centroid[k,n]
    const int n = t;
    float acc = 0.f;
    for (int g = 0; g < GRP; ++g) {
      const _Float16* xg = XE + g * ND + n;
      float s = 0.f;
      for (int r = 0; r < M_TILE; ++r)
        s += A2[r * GRP + g] * (float)xg[r * D2];
      acc += s;
    }
    for (int k = 0; k < NC; ++k) acc -= WK[k] * cen[k * ND + n];
    vpart[((size_t)tile * NB + b) * ND + n] = acc;  // deterministic, no atomics
  }
}

__global__ __launch_bounds__(256) void nextvlad_finalize(
    const float* __restrict__ vpart, float* __restrict__ out) {
  __shared__ float red[256];
  const int b = blockIdx.x, t = threadIdx.x;
  float s = 0.f;
  for (int tile = 0; tile < NTILES; ++tile)
    s += vpart[((size_t)tile * NB + b) * ND + t];
  const float v = s * (1.0f / (float)MSEQ);
  red[t] = v * v;
  __syncthreads();
  for (int k = 128; k > 0; k >>= 1) {
    if (t < k) red[t] += red[t + k];
    __syncthreads();
  }
  const float inv = 1.0f / fmaxf(sqrtf(red[0]), 1e-12f);
  out[b * ND + t] = v * inv;
}

extern "C" void kernel_launch(void* const* d_in, const int* in_sizes, int n_in,
                              void* d_out, int out_size, void* d_ws, size_t ws_size,
                              hipStream_t stream) {
  const float* x   = (const float*)d_in[0];
  // d_in[1] = mask (bool, unused by the reference math)
  const float* We  = (const float*)d_in[2];
  const float* be  = (const float*)d_in[3];
  const float* W1  = (const float*)d_in[4];
  const float* b1  = (const float*)d_in[5];
  const float* W2  = (const float*)d_in[6];
  const float* b2  = (const float*)d_in[7];
  const float* cen = (const float*)d_in[8];
  float* out = (float*)d_out;

  char* ws = (char*)d_ws;
  float*    vpart = (float*)(ws + WS_VPART);
  _Float16* We16  = (_Float16*)(ws + WS_WE16);
  _Float16* W116  = (_Float16*)(ws + WS_W116);

  (void)hipFuncSetAttribute((const void*)nextvlad_main,
                            hipFuncAttributeMaxDynamicSharedMemorySize, LDS_BYTES);

  // Per-launch weight downconvert (idempotent, L2-resident afterwards).
  {
    const int nWe4 = (D2 * D_IN) / 4;      // 524288
    const int nW14 = (GRP * NC * D2) / 4;  // 131072
    cvt_f32_to_f16<<<(nWe4 + 255) / 256, 256, 0, stream>>>(We, We16, nWe4);
    cvt_f32_to_f16<<<(nW14 + 255) / 256, 256, 0, stream>>>(W1, W116, nW14);
  }

  dim3 grid(NTILES, NB);
  nextvlad_main<<<grid, 256, LDS_BYTES, stream>>>(x, We16, be, W116, b1, W2, b2,
                                                  cen, vpart);
  nextvlad_finalize<<<NB, 256, 0, stream>>>(vpart, out);
}
